// Attention_69535520522491
// MI455X (gfx1250) — matrix operand order repaired and checked by reading, not compiled, not run on hardware
//
#include <hip/hip_runtime.h>
#include <hip/hip_bf16.h>

#define NNODES 50000
#define NEDGES 800000
// DIM=128, H=2, C=64

typedef __attribute__((ext_vector_type(8)))  float  v8f;
typedef __attribute__((ext_vector_type(16))) __bf16 v16bf;
typedef __attribute__((ext_vector_type(8)))  __bf16 v8bf;

__device__ __forceinline__ v8f wmma_bf16(v16bf a, v16bf b, v8f c) {
  // D = A(16x32 bf16) x B(32x16 bf16) + C(16x16 f32)
  return __builtin_amdgcn_wmma_f32_16x16x32_bf16(false, a, false, b, (short)0, c,
                                                 false, false);
}

// ---- fragment builders (wave32 layouts per CDNA5 ISA 7.12.2) ----
// A (16x32, 16-bit): lane ln, g=ln>>4, M=ln&15.
//   elems 0..7  -> K = kk + 8g + j
//   elems 8..15 -> K = kk + 16 + 8g + (j-8)
__device__ __forceinline__ v16bf a_frag_from_f32(const float* row, int kk, int g) {
  const float* p0 = row + kk + g * 8;
  const float* p1 = p0 + 16;
  float4 a0 = *(const float4*)(p0);
  float4 a1 = *(const float4*)(p0 + 4);
  float4 b0 = *(const float4*)(p1);
  float4 b1 = *(const float4*)(p1 + 4);
  v16bf f;
  f[0]  = (__bf16)a0.x; f[1]  = (__bf16)a0.y; f[2]  = (__bf16)a0.z; f[3]  = (__bf16)a0.w;
  f[4]  = (__bf16)a1.x; f[5]  = (__bf16)a1.y; f[6]  = (__bf16)a1.z; f[7]  = (__bf16)a1.w;
  f[8]  = (__bf16)b0.x; f[9]  = (__bf16)b0.y; f[10] = (__bf16)b0.z; f[11] = (__bf16)b0.w;
  f[12] = (__bf16)b1.x; f[13] = (__bf16)b1.y; f[14] = (__bf16)b1.z; f[15] = (__bf16)b1.w;
  return f;
}

// B (32x16, 16-bit): lane ln, g=ln>>4, N=ln&15. elems 0..7 -> K = kk + 8g + j, elems 8..15 -> K = kk + 16 + 8g + (j-8).
// Weights pre-transposed to bf16 layout wt[n*128 + k] -> two runs of 8 contiguous K = 16B each.
__device__ __forceinline__ v16bf b_frag_from_bf16(const __bf16* wt, int n, int kk, int g) {
  const __bf16* p = wt + n * 128 + kk + g * 8;
  v8bf lo = *(const v8bf*)(p);
  v8bf hi = *(const v8bf*)(p + 16);
  v16bf f;
#pragma unroll
  for (int i = 0; i < 8; i++) { f[i] = lo[i]; f[i + 8] = hi[i]; }
  return f;
}

// monotone float <-> uint mapping for atomicMax segment-max
__device__ __forceinline__ unsigned fkey(float f) {
  unsigned b = __float_as_uint(f);
  return (b & 0x80000000u) ? ~b : (b | 0x80000000u);
}
__device__ __forceinline__ float fdecode(unsigned u) {
  unsigned b = (u & 0x80000000u) ? (u ^ 0x80000000u) : ~u;
  return __uint_as_float(b);
}

// ---- K0a: transpose + convert all 6 weight matrices to bf16 [n][k] ----
__global__ void prep_weights(const float* Wq, const float* Wk, const float* Wv,
                             const float* Wskip, const float* We, const float* Wproj,
                             __bf16* out) {
  int idx = blockIdx.x * 256 + threadIdx.x;  // exactly 6*16384
  const float* mats[6] = {Wq, Wk, Wv, Wskip, We, Wproj};
  int mat = idx >> 14;
  int i = idx & 16383;
  int n = i >> 7, k = i & 127;
  out[mat * 16384 + n * 128 + k] = (__bf16)mats[mat][k * 128 + n];
}

// ---- K0b: init segment-max keys (-"inf") and denominators ----
__global__ void init_md(unsigned* mkeys, float* denom) {
  int t = blockIdx.x * 256 + threadIdx.x;
  if (t < NNODES * 2) { mkeys[t] = 0u; denom[t] = 0.f; }
}

// ---- K1: q/k/v/skip projections (WMMA). q is pre-scaled by 1/sqrt(C)=0.125.
//      accum is seeded with the skip branch. ----
__global__ void __launch_bounds__(256) node_qkvs(
    const float* __restrict__ x, const __bf16* __restrict__ wbf,
    const float* __restrict__ bq, const float* __restrict__ bk,
    const float* __restrict__ bv, const float* __restrict__ bskip,
    float* __restrict__ qb, float* __restrict__ kb,
    float* __restrict__ vb, float* __restrict__ accum) {
  int tid = threadIdx.x;
  int rowbase = blockIdx.x * 16;
  int wv_ = tid >> 5, ln = tid & 31;
  int g = ln >> 4, nn = ln & 15;
  int colbase = wv_ * 16;
  const __bf16* wq  = wbf;
  const __bf16* wk  = wbf + 16384;
  const __bf16* wvm = wbf + 2 * 16384;
  const __bf16* wsm = wbf + 3 * 16384;
  v8f zero = {0.f, 0.f, 0.f, 0.f, 0.f, 0.f, 0.f, 0.f};
  v8f aq = zero, ak = zero, av = zero, as_ = zero;
  const float* xrow = x + (size_t)(rowbase + nn) * 128;
#pragma unroll
  for (int kk = 0; kk < 128; kk += 32) {
    v16bf a  = a_frag_from_f32(xrow, kk, g);
    v16bf b0 = b_frag_from_bf16(wq,  colbase + nn, kk, g);
    v16bf b1 = b_frag_from_bf16(wk,  colbase + nn, kk, g);
    v16bf b2 = b_frag_from_bf16(wvm, colbase + nn, kk, g);
    v16bf b3 = b_frag_from_bf16(wsm, colbase + nn, kk, g);
    aq  = wmma_bf16(a, b0, aq);
    ak  = wmma_bf16(a, b1, ak);
    av  = wmma_bf16(a, b2, av);
    as_ = wmma_bf16(a, b3, as_);
  }
  int col = colbase + nn;
  float Bq = bq[col], Bk = bk[col], Bv = bv[col], Bs = bskip[col];
#pragma unroll
  for (int r = 0; r < 8; r++) {
    size_t row = (size_t)(rowbase + r + g * 8) * 128 + col;
    qb[row]    = (aq[r] + Bq) * 0.125f;
    kb[row]    = ak[r] + Bk;
    vb[row]    = av[r] + Bv;
    accum[row] = as_[r] + Bs;
  }
}

// ---- K2: fused edge GEMM (e = edge_attr @ We) + alpha + segment max ----
__global__ void __launch_bounds__(128) edge_alpha(
    const float* __restrict__ ea, const __bf16* __restrict__ webf,
    const int* __restrict__ eidx,
    const float* __restrict__ qb, const float* __restrict__ kb,
    float* __restrict__ alpha, unsigned* __restrict__ mkeys) {
  __shared__ float lds[4 * 16 * 129];  // per-wave 16x128 tile of e, padded stride
  int tid = threadIdx.x;
  int wid = tid >> 5, ln = tid & 31;
  int ebase = (blockIdx.x * 4 + wid) * 16;
  int g = ln >> 4, nn = ln & 15;
  v8f zero = {0.f, 0.f, 0.f, 0.f, 0.f, 0.f, 0.f, 0.f};
  v8f acc[8];
#pragma unroll
  for (int t = 0; t < 8; t++) acc[t] = zero;
  const float* erow = ea + (size_t)(ebase + nn) * 128;
#pragma unroll
  for (int kk = 0; kk < 128; kk += 32) {
    v16bf a = a_frag_from_f32(erow, kk, g);
#pragma unroll
    for (int ct = 0; ct < 8; ct++) {
      v16bf b = b_frag_from_bf16(webf, ct * 16 + nn, kk, g);
      acc[ct] = wmma_bf16(a, b, acc[ct]);
    }
  }
  float* myl = lds + wid * (16 * 129);
#pragma unroll
  for (int ct = 0; ct < 8; ct++)
#pragma unroll
    for (int r = 0; r < 8; r++)
      myl[(r + g * 8) * 129 + ct * 16 + nn] = acc[ct][r];
  __syncthreads();
  // thread = (edge-in-tile = nn, head = g)
  int edge = ebase + nn;
  int h = g;
  int sN = eidx[edge];           // src
  int dN = eidx[NEDGES + edge];  // dst
  const float* qr = qb + (size_t)dN * 128 + h * 64;  // already scaled by 1/8
  const float* kr = kb + (size_t)sN * 128 + h * 64;
  const float* er = myl + nn * 129 + h * 64;
  float s = 0.f;
#pragma unroll
  for (int c = 0; c < 64; c += 4) {
    float4 qv = *(const float4*)(qr + c);
    float4 kv = *(const float4*)(kr + c);
    s += qv.x * (kv.x + er[c])     + qv.y * (kv.y + er[c + 1]) +
         qv.z * (kv.z + er[c + 2]) + qv.w * (kv.w + er[c + 3]);
  }
  alpha[edge * 2 + h] = s;
  atomicMax(mkeys + dN * 2 + h, fkey(s));
}

// ---- K3: p = exp(alpha - m[dst]); denom += p (stores p in place) ----
__global__ void softmax_denom(const int* __restrict__ eidx,
                              const unsigned* __restrict__ mkeys,
                              float* __restrict__ alpha,
                              float* __restrict__ denom) {
  int t = blockIdx.x * 256 + threadIdx.x;
  if (t >= NEDGES * 2) return;
  int edge = t >> 1, h = t & 1;
  int dN = eidx[NEDGES + edge];
  float m = fdecode(mkeys[dN * 2 + h]);
  float p = __expf(alpha[t] - m);
  alpha[t] = p;
  atomicAdd(denom + dN * 2 + h, p);
}

// ---- K4: recompute e via WMMA, scatter (v[src]+e)*a into accum ----
__global__ void __launch_bounds__(128) edge_message(
    const float* __restrict__ ea, const __bf16* __restrict__ webf,
    const int* __restrict__ eidx,
    const float* __restrict__ vb, const float* __restrict__ pbuf,
    const float* __restrict__ denom, float* __restrict__ accum) {
  int tid = threadIdx.x;
  int wid = tid >> 5, ln = tid & 31;
  int ebase = (blockIdx.x * 4 + wid) * 16;
  int g = ln >> 4, nn = ln & 15;
  v8f zero = {0.f, 0.f, 0.f, 0.f, 0.f, 0.f, 0.f, 0.f};
  v8f acc[8];
#pragma unroll
  for (int t = 0; t < 8; t++) acc[t] = zero;
  const float* erow = ea + (size_t)(ebase + nn) * 128;
#pragma unroll
  for (int kk = 0; kk < 128; kk += 32) {
    v16bf a = a_frag_from_f32(erow, kk, g);
#pragma unroll
    for (int ct = 0; ct < 8; ct++) {
      v16bf b = b_frag_from_bf16(webf, ct * 16 + nn, kk, g);
      acc[ct] = wmma_bf16(a, b, acc[ct]);
    }
  }
  // D-fragment layout: VGPR r -> edge row (r + 8g), col = 16*ct + nn
#pragma unroll
  for (int rr = 0; rr < 8; rr++) {
    int edge = ebase + rr + g * 8;
    int sN = eidx[edge];
    int dN = eidx[NEDGES + edge];
    float c0 = pbuf[edge * 2 + 0] / denom[dN * 2 + 0];
    float c1 = pbuf[edge * 2 + 1] / denom[dN * 2 + 1];
    const float* vr = vb + (size_t)sN * 128;
    float* ar = accum + (size_t)dN * 128;
#pragma unroll
    for (int ct = 0; ct < 8; ct++) {
      int col = ct * 16 + nn;
      float coef = (ct < 4) ? c0 : c1;  // head = col>=64
      atomicAdd(ar + col, (vr[col] + acc[ct][rr]) * coef);
    }
  }
}

// ---- K5: out = accum @ Wproj + bproj ----
__global__ void __launch_bounds__(256) final_proj(
    const float* __restrict__ accum, const __bf16* __restrict__ wpbf,
    const float* __restrict__ bproj, float* __restrict__ out) {
  int tid = threadIdx.x;
  int rowbase = blockIdx.x * 16;
  int wv_ = tid >> 5, ln = tid & 31;
  int g = ln >> 4, nn = ln & 15;
  int colbase = wv_ * 16;
  v8f d = {0.f, 0.f, 0.f, 0.f, 0.f, 0.f, 0.f, 0.f};
  const float* arow = accum + (size_t)(rowbase + nn) * 128;
#pragma unroll
  for (int kk = 0; kk < 128; kk += 32) {
    v16bf a = a_frag_from_f32(arow, kk, g);
    v16bf b = b_frag_from_bf16(wpbf, colbase + nn, kk, g);
    d = wmma_bf16(a, b, d);
  }
  int col = colbase + nn;
  float B = bproj[col];
#pragma unroll
  for (int r = 0; r < 8; r++)
    out[(size_t)(rowbase + r + g * 8) * 128 + col] = d[r] + B;
}

extern "C" void kernel_launch(void* const* d_in, const int* in_sizes, int n_in,
                              void* d_out, int out_size, void* d_ws, size_t ws_size,
                              hipStream_t stream) {
  const float* x     = (const float*)d_in[0];
  const float* ea    = (const float*)d_in[1];
  const float* Wq    = (const float*)d_in[2];
  const float* bq    = (const float*)d_in[3];
  const float* Wk    = (const float*)d_in[4];
  const float* bk    = (const float*)d_in[5];
  const float* Wv    = (const float*)d_in[6];
  const float* bv    = (const float*)d_in[7];
  const float* We    = (const float*)d_in[8];
  const float* Wskip = (const float*)d_in[9];
  const float* bskip = (const float*)d_in[10];
  const float* Wproj = (const float*)d_in[11];
  const float* bproj = (const float*)d_in[12];
  const int*   eidx  = (const int*)d_in[13];
  float* outp = (float*)d_out;

  char* ws = (char*)d_ws;
  size_t off = 0;
  float* qb       = (float*)(ws + off); off += (size_t)NNODES * 128 * 4;
  float* kb       = (float*)(ws + off); off += (size_t)NNODES * 128 * 4;
  float* vbuf     = (float*)(ws + off); off += (size_t)NNODES * 128 * 4;
  float* accum    = (float*)(ws + off); off += (size_t)NNODES * 128 * 4;
  float* alpha    = (float*)(ws + off); off += (size_t)NEDGES * 2 * 4;
  unsigned* mkeys = (unsigned*)(ws + off); off += (size_t)NNODES * 2 * 4;
  float* denom    = (float*)(ws + off); off += (size_t)NNODES * 2 * 4;
  __bf16* wbf     = (__bf16*)(ws + off); off += 6 * 16384 * 2;

  prep_weights<<<384, 256, 0, stream>>>(Wq, Wk, Wv, Wskip, We, Wproj, wbf);
  init_md<<<(NNODES * 2 + 255) / 256, 256, 0, stream>>>(mkeys, denom);
  node_qkvs<<<NNODES / 16, 256, 0, stream>>>(x, wbf, bq, bk, bv, bskip,
                                             qb, kb, vbuf, accum);
  edge_alpha<<<NEDGES / 64, 128, 0, stream>>>(ea, wbf + 4 * 16384, eidx,
                                              qb, kb, alpha, mkeys);
  softmax_denom<<<(NEDGES * 2 + 255) / 256, 256, 0, stream>>>(eidx, mkeys,
                                                              alpha, denom);
  edge_message<<<NEDGES / 64, 128, 0, stream>>>(ea, wbf + 4 * 16384, eidx,
                                                vbuf, alpha, denom, accum);
  final_proj<<<NNODES / 16, 256, 0, stream>>>(accum, wbf + 5 * 16384, bproj, outp);
}
